// Multi_Head_Self_Attention_6305011990836
// MI455X (gfx1250) — matrix operand order repaired and checked by reading, not compiled, not run on hardware
//
#include <hip/hip_runtime.h>
#include <hip/hip_bf16.h>
#include <stdint.h>

// MI455X / gfx1250, wave32, WMMA bf16 path + async global->LDS staging.
// B=4, S=2048, D=1024, H=16, Dh=64

#define DIMD   1024
#define NH     16
#define DH     64
#define SEQ    2048
#define BATCH  4

typedef __attribute__((ext_vector_type(16))) __bf16 v16bf;
typedef __attribute__((ext_vector_type(8)))  float  v8f;

union FragU { unsigned u[8]; v16bf v; };

// cheap round-to-nearest (half-up) f32->bf16
__device__ __forceinline__ unsigned short f2bf(float f) {
  union { float f; unsigned u; } x; x.f = f;
  return (unsigned short)((x.u + 0x8000u) >> 16);
}

// pack two f32 -> dword of two bf16 (lo in [15:0]) via one v_perm_b32
__device__ __forceinline__ unsigned packbf2(float lo, float hi) {
  union { float f; unsigned u; } a, b;
  a.f = lo; b.f = hi;
  return __builtin_amdgcn_perm(b.u + 0x8000u, a.u + 0x8000u, 0x07060302u);
}

// ---- async global->LDS (CDNA5 GLOBAL_LOAD_ASYNC_TO_LDS_B128, ASYNCcnt) ----
// Flat LDS pointers carry the LDS byte offset in their low 32 bits
__device__ __forceinline__ unsigned lds_addr_of(const void* p) {
  return (unsigned)(size_t)p;
}
__device__ __forceinline__ void async_copy_b128(unsigned lds_dst, const void* gsrc) {
  asm volatile("global_load_async_to_lds_b128 %0, %1, off"
               :
               : "v"(lds_dst), "v"((unsigned long long)(size_t)gsrc)
               : "memory");
}
__device__ __forceinline__ void wait_async() {
  asm volatile("s_wait_asynccnt 0" ::: "memory");
}

// A operand (16xK=32, bf16).  lane: M = lane&15, half = lane>>4.
// element v (dword): K = 2*(v&3) + 16*(v>>2) + 8*half  (pairs contiguous in memory)
__device__ __forceinline__ v16bf loadA(const unsigned short* p, int RS, int row0,
                                       int k_off, int lane) {
  const int r = row0 + (lane & 15);
  const int h = lane >> 4;
  const unsigned short* q = p + r * RS + k_off + h * 8;
  FragU f;
#pragma unroll
  for (int v = 0; v < 8; ++v) {
    const int kk = ((v & 3) * 2) + ((v >> 2) * 16);
    f.u[v] = *(const unsigned*)(q + kk);
  }
  return f.v;
}

// B operand (K=32 x 16, bf16).  lane: N = lane&15, half = lane>>4.
// element v (dword): K = 2*(v&3) + 16*(v>>2) + 8*half.  Tile stored [N rows][K cols].
__device__ __forceinline__ v16bf loadB(const unsigned short* p, int RS, int row0,
                                       int k_off, int lane) {
  const int r = row0 + (lane & 15);
  const int h = lane >> 4;
  const unsigned short* q = p + r * RS + k_off + h * 8;
  FragU f;
#pragma unroll
  for (int v = 0; v < 8; ++v) f.u[v] = *(const unsigned*)(q + ((v & 3) * 2) + ((v >> 2) * 16));
  return f.v;
}

__device__ __forceinline__ v8f wmma_bf16(v16bf a, v16bf b, v8f c) {
  return __builtin_amdgcn_wmma_f32_16x16x32_bf16(false, a, false, b, (short)0, c,
                                                 false, false);
}

// ---------------------------------------------------------------------------
// Kernel 1: projection GEMM.  out[b,h,s,dh] (bf16) = X[b*S+s,:] . W[o,:] + bias[o]
// Block tile: 256(M) x 64(N), K-step 32.  8 waves; wave computes 32 x 64.
// (f32->bf16 conversion happens in-register, so staging keeps the VGPR path.)
// ---------------------------------------------------------------------------
__global__ __launch_bounds__(256) void qkv_proj_kernel(
    const float* __restrict__ X, const float* __restrict__ W,
    const float* __restrict__ bias, unsigned short* __restrict__ out) {
  __shared__ unsigned short Xs[2][256 * 36];  // 32-col rows, stride 36 (8B aligned)
  __shared__ unsigned short Ws[2][64 * 36];

  const int t = threadIdx.x;
  const int lane = t & 31, wave = t >> 5;
  const int col = lane & 15, half = lane >> 4;
  const int m_base = blockIdx.y * 256;
  const int n_base = blockIdx.x * 64;

  v8f acc[2][4];
#pragma unroll
  for (int mt = 0; mt < 2; ++mt)
#pragma unroll
    for (int nc = 0; nc < 4; ++nc)
#pragma unroll
      for (int i = 0; i < 8; ++i) acc[mt][nc][i] = 0.0f;

  float bval[4];
#pragma unroll
  for (int nc = 0; nc < 4; ++nc) bval[nc] = bias[n_base + nc * 16 + col];

  float4 xr[8];
  float4 wr[2];

  auto fetch = [&](int k0) {
#pragma unroll
    for (int q = 0; q < 8; ++q) {
      int i = t + q * 256, row = i >> 3, c4 = i & 7;
      xr[q] = *(const float4*)(X + (size_t)(m_base + row) * DIMD + k0 + c4 * 4);
    }
#pragma unroll
    for (int q = 0; q < 2; ++q) {
      int i = t + q * 256, row = i >> 3, c4 = i & 7;
      wr[q] = *(const float4*)(W + (size_t)(n_base + row) * DIMD + k0 + c4 * 4);
    }
  };
  auto stage = [&](int buf) {
#pragma unroll
    for (int q = 0; q < 8; ++q) {
      int i = t + q * 256, row = i >> 3, c4 = i & 7;
      uint2 pk;
      pk.x = packbf2(xr[q].x, xr[q].y);
      pk.y = packbf2(xr[q].z, xr[q].w);
      *(uint2*)(&Xs[buf][row * 36 + c4 * 4]) = pk;
    }
#pragma unroll
    for (int q = 0; q < 2; ++q) {
      int i = t + q * 256, row = i >> 3, c4 = i & 7;
      uint2 pk;
      pk.x = packbf2(wr[q].x, wr[q].y);
      pk.y = packbf2(wr[q].z, wr[q].w);
      *(uint2*)(&Ws[buf][row * 36 + c4 * 4]) = pk;
    }
  };

  fetch(0);
  stage(0);
  __syncthreads();

  int buf = 0;
  for (int k = 0; k < 32; ++k) {
    if (k + 1 < 32) fetch((k + 1) * 32);
    v16bf aF0 = loadA(&Xs[buf][0], 36, wave * 32, 0, lane);
    v16bf aF1 = loadA(&Xs[buf][0], 36, wave * 32 + 16, 0, lane);
#pragma unroll
    for (int nc = 0; nc < 4; ++nc) {
      v16bf bF = loadB(&Ws[buf][0], 36, nc * 16, 0, lane);
      acc[0][nc] = wmma_bf16(aF0, bF, acc[0][nc]);
      acc[1][nc] = wmma_bf16(aF1, bF, acc[1][nc]);
    }
    if (k + 1 < 32) stage(buf ^ 1);
    __syncthreads();
    buf ^= 1;
  }

  // epilogue: bias add, bf16 store to head-split layout [B,H,S,Dh]
#pragma unroll
  for (int mt = 0; mt < 2; ++mt)
#pragma unroll
    for (int nc = 0; nc < 4; ++nc) {
      const int o = n_base + nc * 16 + col;
      const int hh = o >> 6, dh = o & 63;
#pragma unroll
      for (int i = 0; i < 8; ++i) {
        const int m = m_base + wave * 32 + mt * 16 + i + half * 8;
        const int b = m >> 11, s = m & 2047;
        out[((size_t)((b * NH + hh) * SEQ + s)) * DH + dh] =
            f2bf(acc[mt][nc][i] + bval[nc]);
      }
    }
}

// ---------------------------------------------------------------------------
// Kernel 2: flash attention.  Grid: (S/256, B*H).  8 waves; wave owns 32 q-rows.
// Q and K tiles staged with GLOBAL_LOAD_ASYNC_TO_LDS_B128 (ASYNCcnt);
// V staged through registers because it is transposed into LDS.
// ---------------------------------------------------------------------------
__global__ __launch_bounds__(256) void attn_kernel(
    const unsigned short* __restrict__ qp, const unsigned short* __restrict__ kp,
    const unsigned short* __restrict__ vp, const float* __restrict__ mask,
    float* __restrict__ out) {
  __shared__ unsigned short Qs[256 * 72];      // 64-col rows, stride 72 (16B aligned)
  __shared__ unsigned short Ks[2][32 * 72];    // [kv row][dh]
  __shared__ unsigned short Vt[2][64 * 34];    // transposed: [dh][kv]
  __shared__ float Msk[2][32];
  __shared__ unsigned short Ps[8][16 * 34];    // per-wave P tile (A-layout source)

  const int t = threadIdx.x, lane = t & 31, wave = t >> 5;
  const int col = lane & 15, half = lane >> 4;
  const int bh = blockIdx.y;
  const int b = bh >> 4, h = bh & 15;
  const int qblk = blockIdx.x;

  const unsigned short* qb = qp + (size_t)bh * SEQ * DH;
  const unsigned short* kb = kp + (size_t)bh * SEQ * DH;
  const unsigned short* vb = vp + (size_t)bh * SEQ * DH;
  const float* mb = mask + b * SEQ;

  // stage Q block (256 x 64 bf16): 2048 async b128 copies, 8 per thread
#pragma unroll
  for (int q = 0; q < 8; ++q) {
    int i = t + q * 256, row = i >> 3, c = (i & 7) * 8;
    async_copy_b128(lds_addr_of(&Qs[row * 72 + c]),
                    qb + (size_t)(qblk * 256 + row) * DH + c);
  }

  const int krow = t >> 3, kc = (t & 7) * 8;
  uint4 vreg = {};
  float mreg = 0.0f;

  auto fetchKV = [&](int kv0, int buf) {
    // K tile: straight copy -> async global->LDS, lands in Ks[buf]
    async_copy_b128(lds_addr_of(&Ks[buf][krow * 72 + kc]),
                    kb + (size_t)(kv0 + krow) * DH + kc);
    // V tile needs a transpose -> register path
    vreg = *(const uint4*)(vb + (size_t)(kv0 + krow) * DH + kc);
    if (t < 32) mreg = mb[kv0 + t];
  };
  auto stageV = [&](int buf) {
    union { uint4 u; unsigned short s[8]; } vu;
    vu.u = vreg;
#pragma unroll
    for (int j = 0; j < 8; ++j) Vt[buf][(kc + j) * 34 + krow] = vu.s[j];
    if (t < 32) Msk[buf][t] = mreg;
  };

  fetchKV(0, 0);
  stageV(0);
  wait_async();        // Q block + first K tile resident
  __syncthreads();

  // Q fragments: two M tiles x two K=32 chunks, loaded once
  v16bf aQ[2][2];
#pragma unroll
  for (int mt = 0; mt < 2; ++mt) {
    aQ[mt][0] = loadA(Qs, 72, wave * 32 + mt * 16, 0, lane);
    aQ[mt][1] = loadA(Qs, 72, wave * 32 + mt * 16, 32, lane);
  }

  v8f accO[2][4];
#pragma unroll
  for (int mt = 0; mt < 2; ++mt)
#pragma unroll
    for (int nc = 0; nc < 4; ++nc)
#pragma unroll
      for (int i = 0; i < 8; ++i) accO[mt][nc][i] = 0.0f;

  float mrun[2][8], lrun[2][8];
#pragma unroll
  for (int mt = 0; mt < 2; ++mt)
#pragma unroll
    for (int i = 0; i < 8; ++i) { mrun[mt][i] = -3.0e38f; lrun[mt][i] = 0.0f; }

  const float scale = 0.125f;  // 1/sqrt(64)
  int buf = 0;

  for (int step = 0; step < 64; ++step) {
    if (step + 1 < 64) fetchKV((step + 1) * 32, buf ^ 1);  // async K -> idle buffer

    // K B-fragments (shared by both M tiles)
    v16bf bk00 = loadB(Ks[buf], 72, 0, 0, lane);
    v16bf bk01 = loadB(Ks[buf], 72, 0, 32, lane);
    v16bf bk10 = loadB(Ks[buf], 72, 16, 0, lane);
    v16bf bk11 = loadB(Ks[buf], 72, 16, 32, lane);
    // V B-fragments (shared by both M tiles)
    v16bf bV[4];
#pragma unroll
    for (int nc = 0; nc < 4; ++nc) bV[nc] = loadB(Vt[buf], 34, nc * 16, 0, lane);

    const float madd0 = -10000.0f * (1.0f - Msk[buf][col]);
    const float madd1 = -10000.0f * (1.0f - Msk[buf][16 + col]);

#pragma unroll
    for (int mt = 0; mt < 2; ++mt) {
      v8f c0, c1;
#pragma unroll
      for (int i = 0; i < 8; ++i) { c0[i] = 0.0f; c1[i] = 0.0f; }
      c0 = wmma_bf16(aQ[mt][0], bk00, c0);
      c0 = wmma_bf16(aQ[mt][1], bk01, c0);
      c1 = wmma_bf16(aQ[mt][0], bk10, c1);
      c1 = wmma_bf16(aQ[mt][1], bk11, c1);

      // online softmax; C-layout row i lives at lanes sharing (lane>>4)
#pragma unroll
      for (int i = 0; i < 8; ++i) {
        float s0 = c0[i] * scale + madd0;
        float s1 = c1[i] * scale + madd1;
        float mx = fmaxf(s0, s1);
#pragma unroll
        for (int off = 1; off < 16; off <<= 1)
          mx = fmaxf(mx, __shfl_xor(mx, off, 32));
        const float mnew = fmaxf(mrun[mt][i], mx);
        const float alpha = __expf(mrun[mt][i] - mnew);
        mrun[mt][i] = mnew;
        const float p0 = __expf(s0 - mnew);
        const float p1 = __expf(s1 - mnew);
        float rs = p0 + p1;
#pragma unroll
        for (int off = 1; off < 16; off <<= 1) rs += __shfl_xor(rs, off, 32);
        lrun[mt][i] = lrun[mt][i] * alpha + rs;
#pragma unroll
        for (int nc = 0; nc < 4; ++nc) accO[mt][nc][i] *= alpha;
        const int row = i + half * 8;
        Ps[wave][row * 34 + col] = f2bf(p0);
        Ps[wave][row * 34 + 16 + col] = f2bf(p1);
      }

      // O += P . V   (per-wave Ps buffer; LDS in-order within a wave)
      v16bf pA = loadA(&Ps[wave][0], 34, 0, 0, lane);
#pragma unroll
      for (int nc = 0; nc < 4; ++nc)
        accO[mt][nc] = wmma_bf16(pA, bV[nc], accO[mt][nc]);
    }

    if (step + 1 < 64) stageV(buf ^ 1);
    wait_async();      // this wave's async K stores for buf^1 are in LDS
    __syncthreads();   // publish buf^1 to all waves
    buf ^= 1;
  }

  // epilogue: normalize by row sum, merge heads -> out[b, s, h*64+dh] f32
#pragma unroll
  for (int mt = 0; mt < 2; ++mt)
#pragma unroll
    for (int i = 0; i < 8; ++i) {
      const float invl = 1.0f / lrun[mt][i];
      const int s = qblk * 256 + wave * 32 + mt * 16 + i + half * 8;
#pragma unroll
      for (int nc = 0; nc < 4; ++nc) {
        const int dh = nc * 16 + col;
        out[((size_t)(b * SEQ + s)) * DIMD + h * DH + dh] = accO[mt][nc][i] * invl;
      }
    }
}

// ---------------------------------------------------------------------------
extern "C" void kernel_launch(void* const* d_in, const int* in_sizes, int n_in,
                              void* d_out, int out_size, void* d_ws, size_t ws_size,
                              hipStream_t stream) {
  const float* q    = (const float*)d_in[0];
  const float* k    = (const float*)d_in[1];
  const float* v    = (const float*)d_in[2];
  const float* mask = (const float*)d_in[3];
  const float* wq   = (const float*)d_in[4];
  const float* bq   = (const float*)d_in[5];
  const float* wk   = (const float*)d_in[6];
  const float* bk   = (const float*)d_in[7];
  const float* wv   = (const float*)d_in[8];
  const float* bv   = (const float*)d_in[9];

  // workspace: bf16 projected Q/K/V in [B,H,S,Dh] layout (16 MB each)
  unsigned short* qp = (unsigned short*)d_ws;
  unsigned short* kp = qp + (size_t)BATCH * SEQ * DIMD;
  unsigned short* vp = kp + (size_t)BATCH * SEQ * DIMD;

  dim3 gproj(DIMD / 64, (BATCH * SEQ) / 256);
  qkv_proj_kernel<<<gproj, 256, 0, stream>>>(q, wq, bq, qp);
  qkv_proj_kernel<<<gproj, 256, 0, stream>>>(k, wk, bk, kp);
  qkv_proj_kernel<<<gproj, 256, 0, stream>>>(v, wv, bv, vp);

  dim3 gattn(SEQ / 256, BATCH * NH);
  attn_kernel<<<gattn, 256, 0, stream>>>(qp, kp, vp, mask, (float*)d_out);
}
